// MultiHeadAttention_28819230556540
// MI455X (gfx1250) — matrix operand order repaired and checked by reading, not compiled, not run on hardware
//
#include <hip/hip_runtime.h>
#include <hip/hip_bf16.h>
#include <math.h>

// ---------------------------------------------------------------------------
// Problem constants (reference: B=4, S=2048, D=1024, H=16, Hd=64)
// ---------------------------------------------------------------------------
constexpr int kB  = 4;
constexpr int kS  = 2048;
constexpr int kD  = 1024;
constexpr int kH  = 16;
constexpr int kHd = 64;
constexpr int kM  = kB * kS;          // 8192 rows for the projection GEMMs

typedef __bf16 bf16;
typedef __attribute__((ext_vector_type(16))) __bf16 v16bf;
typedef __attribute__((ext_vector_type(8)))  __bf16 v8bf;
typedef __attribute__((ext_vector_type(4)))  __bf16 v4bf;
typedef __attribute__((ext_vector_type(8)))  float  v8f;
typedef int v4i __attribute__((vector_size(4 * sizeof(int))));  // matches builtin

// ---------------------------------------------------------------------------
// CDNA5 async global->LDS copy (ASYNCcnt path), with sync fallback.
// ---------------------------------------------------------------------------
#if defined(__gfx1250__) &&                                                   \
    __has_builtin(__builtin_amdgcn_global_load_async_to_lds_b128) &&          \
    __has_builtin(__builtin_amdgcn_s_wait_asynccnt)
#define USE_ASYNC 1
#else
#define USE_ASYNC 0
#endif

static __device__ __forceinline__ void cp16(bf16* ldst, const bf16* gsrc) {
#if USE_ASYNC
  __builtin_amdgcn_global_load_async_to_lds_b128(
      (__attribute__((address_space(1))) v4i*)gsrc,
      (__attribute__((address_space(3))) v4i*)ldst,
      /*offset=*/0, /*cpol=*/0);
#else
  *(v8bf*)ldst = *(const v8bf*)gsrc;
#endif
}

template <int N>
static __device__ __forceinline__ void async_wait() {
#if USE_ASYNC
  __builtin_amdgcn_s_wait_asynccnt(N);
#endif
}

static __device__ __forceinline__ v16bf cat16(v8bf lo, v8bf hi) {
  return __builtin_shufflevector(lo, hi, 0,1,2,3,4,5,6,7,8,9,10,11,12,13,14,15);
}

static __device__ __forceinline__ v8f wmma_bf16(v16bf a, v16bf b, v8f c) {
  // D = A(16x32 bf16) * B(32x16 bf16) + C(16x16 f32)
  return __builtin_amdgcn_wmma_f32_16x16x32_bf16(
      /*neg_a=*/false, a, /*neg_b=*/false, b,
      /*c_mod=*/(short)0, c, /*reuse_a=*/false, /*reuse_b=*/false);
}

// ---------------------------------------------------------------------------
// 1) f32 -> bf16 conversion (vectorized x4)
// ---------------------------------------------------------------------------
__global__ __launch_bounds__(256) void cvt_f32_bf16(const float* __restrict__ in,
                                                    bf16* __restrict__ out, int n4) {
  int i = blockIdx.x * 256 + threadIdx.x;
  if (i < n4) {
    float4 f = ((const float4*)in)[i];
    v4bf o;
    o.x = (bf16)f.x; o.y = (bf16)f.y; o.z = (bf16)f.z; o.w = (bf16)f.w;
    ((v4bf*)out)[i] = o;
  }
}

// ---------------------------------------------------------------------------
// 2) GEMM  C[M x 1024] = Xb[M x 1024] * Wb^T   (Wb is N x K row-major)
//    Block tile 128x128, 8 waves, each wave owns a 16x128 strip.
//    Double-buffered async global->LDS staging of 128x32 tiles.
//    MODE 1: scatter bf16 into (B,H,S,Hd)      (Q / K)
//    MODE 2: scatter bf16 into (B,H,Hd,S)      (V transposed)
//    MODE 3: f32 row-major out                  (final projection)
// ---------------------------------------------------------------------------
constexpr int GSTR = 40;   // LDS row stride in bf16 elems (32 + 8 pad, 80B rows)

template <int MODE>
__global__ __launch_bounds__(256) void gemm_xwt(const bf16* __restrict__ X,
                                                const bf16* __restrict__ W,
                                                void* __restrict__ out) {
  __shared__ alignas(16) bf16 lx[2][128 * GSTR];
  __shared__ alignas(16) bf16 lw[2][128 * GSTR];

  const int tid  = threadIdx.x;
  const int lane = tid & 31;
  const int wave = tid >> 5;
  const int nn   = lane & 15;   // fragment column / row index
  const int hh   = lane >> 4;   // fragment half
  const int m0   = blockIdx.x * 128;
  const int n0   = blockIdx.y * 128;

  const int lrow = tid >> 1;          // 0..127: cooperative tile-load row
  const int lcol = (tid & 1) * 16;    // 0 or 16

  // Stage one 128x32 X tile and one 128x32 W tile into LDS buffer `buf`.
  auto stage = [&](int k0, int buf) {
    const bf16* gx = X + (size_t)(m0 + lrow) * kD + k0 + lcol;
    cp16(lx[buf] + lrow * GSTR + lcol,     gx);
    cp16(lx[buf] + lrow * GSTR + lcol + 8, gx + 8);
    const bf16* gw = W + (size_t)(n0 + lrow) * kD + k0 + lcol;
    cp16(lw[buf] + lrow * GSTR + lcol,     gw);
    cp16(lw[buf] + lrow * GSTR + lcol + 8, gw + 8);
  };

  v8f acc[8];
  const v8f vzero = {0.f,0.f,0.f,0.f,0.f,0.f,0.f,0.f};
#pragma unroll
  for (int i = 0; i < 8; ++i) acc[i] = vzero;

  constexpr int NT = kD / 32;   // 32 k-steps
  stage(0, 0);
  for (int kt = 0; kt < NT; ++kt) {
    const int cur = kt & 1;
    if (kt + 1 < NT) {        // prefetch next tile into the other buffer
      stage((kt + 1) * 32, cur ^ 1);
      async_wait<4>();        // this wave's 4 loads of tile `kt` have landed
    } else {
      async_wait<0>();
    }
    __syncthreads();          // tile `kt` visible to whole block

    const bf16* lxc = lx[cur];
    const bf16* lwc = lw[cur];
    // A fragment: 16 rows (this wave's strip) x 32 K
    const bf16* arow = lxc + (wave * 16 + nn) * GSTR;
    v16bf afrag = cat16(*(const v8bf*)(arow + hh * 8),
                        *(const v8bf*)(arow + 16 + hh * 8));
#pragma unroll
    for (int nt = 0; nt < 8; ++nt) {
      const bf16* brow = lwc + (nt * 16 + nn) * GSTR;
      v16bf bfrag = cat16(*(const v8bf*)(brow + hh * 8),
                          *(const v8bf*)(brow + 16 + hh * 8));
      acc[nt] = wmma_bf16(afrag, bfrag, acc[nt]);
    }
    __syncthreads();          // all waves done reading before buffer reuse
  }

  // Epilogue: C-fragment element (i, lane) is (M = i + 8*hh, N = nn)
#pragma unroll
  for (int nt = 0; nt < 8; ++nt) {
#pragma unroll
    for (int i = 0; i < 8; ++i) {
      const int m = m0 + wave * 16 + i + 8 * hh;
      const int n = n0 + nt * 16 + nn;
      const float v = acc[nt][i];
      if constexpr (MODE == 1) {            // (B,H,S,Hd) bf16
        const int b = m >> 11, s = m & (kS - 1);
        const int h = n >> 6,  d = n & (kHd - 1);
        ((bf16*)out)[(((size_t)b * kH + h) * kS + s) * kHd + d] = (bf16)v;
      } else if constexpr (MODE == 2) {     // (B,H,Hd,S) bf16 (V transposed)
        const int b = m >> 11, s = m & (kS - 1);
        const int h = n >> 6,  d = n & (kHd - 1);
        ((bf16*)out)[(((size_t)b * kH + h) * kHd + d) * kS + s] = (bf16)v;
      } else {                              // f32 row-major
        ((float*)out)[(size_t)m * kD + n] = v;
      }
    }
  }
}

// ---------------------------------------------------------------------------
// 3) Causal flash attention. Grid: (S/128, B*H). Block: 256 threads (8 waves).
//    Each wave owns 16 query rows. K/V double-buffered in LDS per 64-key tile.
// ---------------------------------------------------------------------------
constexpr int FSTR = 72;   // LDS stride (64 + 8 pad, 144B rows)

__global__ __launch_bounds__(256) void flash_attn(const bf16* __restrict__ Q,
                                                  const bf16* __restrict__ K,
                                                  const bf16* __restrict__ Vt,
                                                  bf16* __restrict__ Ctx) {
  __shared__ alignas(16) bf16 lk[2][64 * FSTR];     // 64 keys x 64 d
  __shared__ alignas(16) bf16 lv[2][64 * FSTR];     // 64 d    x 64 keys
  __shared__ alignas(16) bf16 lp[8 * 16 * FSTR];    // per-wave P tiles 16 x 64

  const int bh   = blockIdx.y;                 // b*16 + h
  const int b    = bh >> 4, h = bh & 15;
  const int q0   = blockIdx.x * 128;
  const int tid  = threadIdx.x;
  const int lane = tid & 31;
  const int wave = tid >> 5;
  const int nn   = lane & 15;
  const int hh   = lane >> 4;
  const int qbase = q0 + wave * 16;

  // Q fragments for this wave's 16 rows, both 32-wide d k-steps (held in regs).
  const bf16* qrow = Q + ((size_t)bh * kS + qbase + nn) * kHd;
  v16bf qf[2];
#pragma unroll
  for (int ks = 0; ks < 2; ++ks)
    qf[ks] = cat16(*(const v8bf*)(qrow + ks * 32 + hh * 8),
                   *(const v8bf*)(qrow + ks * 32 + 16 + hh * 8));

  const v8f vzero = {0.f,0.f,0.f,0.f,0.f,0.f,0.f,0.f};
  v8f o[4];
#pragma unroll
  for (int dt = 0; dt < 4; ++dt) o[dt] = vzero;
  float rowm[8], rowl[8];
#pragma unroll
  for (int i = 0; i < 8; ++i) { rowm[i] = -1e30f; rowl[i] = 0.f; }

  const int trow = tid >> 2;            // 0..63 cooperative tile-load row
  const int tcol = (tid & 3) * 16;      // 0/16/32/48
  bf16* pw = lp + wave * 16 * FSTR;     // this wave's private P tile

  // Stage one 64-key K tile and matching V^T tile into buffer `buf`.
  auto stage = [&](int t0, int buf) {
    const bf16* kg = K + ((size_t)bh * kS + t0 + trow) * kHd + tcol;
    cp16(lk[buf] + trow * FSTR + tcol,     kg);
    cp16(lk[buf] + trow * FSTR + tcol + 8, kg + 8);
    const bf16* vg = Vt + ((size_t)bh * kHd + trow) * kS + t0 + tcol;
    cp16(lv[buf] + trow * FSTR + tcol,     vg);
    cp16(lv[buf] + trow * FSTR + tcol + 8, vg + 8);
  };

  const int ntile = (q0 + 128) / 64;    // causal: keys <= last query of strip
  stage(0, 0);
  for (int ti = 0; ti < ntile; ++ti) {
    const int cur = ti & 1;
    if (ti + 1 < ntile) {
      stage((ti + 1) * 64, cur ^ 1);
      async_wait<4>();
    } else {
      async_wait<0>();
    }
    __syncthreads();
    const bf16* lkc = lk[cur];
    const bf16* lvc = lv[cur];
    const int t0 = ti * 64;

    // ---- S = Q K^T over this 64-key tile (4 subtiles x 2 k-steps) ----
    v8f st[4];
#pragma unroll
    for (int t = 0; t < 4; ++t) {
      v8f a = vzero;
#pragma unroll
      for (int ks = 0; ks < 2; ++ks) {
        const bf16* brow = lkc + (t * 16 + nn) * FSTR + ks * 32;
        v16bf bfrag = cat16(*(const v8bf*)(brow + hh * 8),
                            *(const v8bf*)(brow + 16 + hh * 8));
        a = wmma_bf16(qf[ks], bfrag, a);
      }
      st[t] = a;
    }

    // ---- scale + causal mask + per-row tile max ----
    float tmax[8];
#pragma unroll
    for (int i = 0; i < 8; ++i) tmax[i] = -1e30f;
#pragma unroll
    for (int t = 0; t < 4; ++t) {
      const int key = t0 + t * 16 + nn;
#pragma unroll
      for (int i = 0; i < 8; ++i) {
        const int qg = qbase + i + 8 * hh;
        float v = st[t][i] * 0.125f;               // 1/sqrt(64)
        v = (key <= qg) ? v : -1e30f;
        st[t][i] = v;
        tmax[i] = fmaxf(tmax[i], v);
      }
    }
#pragma unroll
    for (int i = 0; i < 8; ++i) {                  // reduce across 16 lanes
      float t = tmax[i];
      t = fmaxf(t, __shfl_xor(t, 1, 32));
      t = fmaxf(t, __shfl_xor(t, 2, 32));
      t = fmaxf(t, __shfl_xor(t, 4, 32));
      t = fmaxf(t, __shfl_xor(t, 8, 32));
      tmax[i] = t;
    }

    // ---- online softmax update ----
    float alpha[8], rsum[8];
#pragma unroll
    for (int i = 0; i < 8; ++i) {
      const float nm = fmaxf(rowm[i], tmax[i]);
      alpha[i] = __expf(rowm[i] - nm);
      rowm[i] = nm;
      rsum[i] = 0.f;
    }
#pragma unroll
    for (int t = 0; t < 4; ++t)
#pragma unroll
      for (int i = 0; i < 8; ++i) {
        const float p = __expf(st[t][i] - rowm[i]);
        st[t][i] = p;
        rsum[i] += p;
      }
#pragma unroll
    for (int i = 0; i < 8; ++i) {
      float r = rsum[i];
      r += __shfl_xor(r, 1, 32);
      r += __shfl_xor(r, 2, 32);
      r += __shfl_xor(r, 4, 32);
      r += __shfl_xor(r, 8, 32);
      rowl[i] = rowl[i] * alpha[i] + r;
    }
#pragma unroll
    for (int dt = 0; dt < 4; ++dt)
#pragma unroll
      for (int i = 0; i < 8; ++i) o[dt][i] *= alpha[i];

    // ---- P (bf16) -> wave-private LDS, re-read as A fragments ----
#pragma unroll
    for (int t = 0; t < 4; ++t)
#pragma unroll
      for (int i = 0; i < 8; ++i)
        pw[(i + 8 * hh) * FSTR + t * 16 + nn] = (bf16)st[t][i];

    // ---- O += P * V (contraction over 64 keys: 2 k-steps x 4 d-subtiles) ----
#pragma unroll
    for (int ks = 0; ks < 2; ++ks) {
      const bf16* arow = pw + nn * FSTR + ks * 32;
      v16bf afrag = cat16(*(const v8bf*)(arow + hh * 8),
                          *(const v8bf*)(arow + 16 + hh * 8));
#pragma unroll
      for (int dt = 0; dt < 4; ++dt) {
        const bf16* brow = lvc + (dt * 16 + nn) * FSTR + ks * 32;
        v16bf bfrag = cat16(*(const v8bf*)(brow + hh * 8),
                            *(const v8bf*)(brow + 16 + hh * 8));
        o[dt] = wmma_bf16(afrag, bfrag, o[dt]);
      }
    }
    __syncthreads();          // all waves done with lk/lv[cur] before refill
  }

  // ---- normalize and write context (B,S,D) as bf16 ----
#pragma unroll
  for (int dt = 0; dt < 4; ++dt)
#pragma unroll
    for (int i = 0; i < 8; ++i) {
      const int qg = qbase + i + 8 * hh;
      const float v = o[dt][i] / rowl[i];
      Ctx[((size_t)b * kS + qg) * kD + h * kHd + dt * 16 + nn] = (bf16)v;
    }
}

// ---------------------------------------------------------------------------
// Host-side orchestration
// ---------------------------------------------------------------------------
extern "C" void kernel_launch(void* const* d_in, const int* in_sizes, int n_in,
                              void* d_out, int out_size, void* d_ws, size_t ws_size,
                              hipStream_t stream) {
  (void)in_sizes; (void)n_in; (void)out_size; (void)ws_size;
  const float* X  = (const float*)d_in[0];
  const float* Wq = (const float*)d_in[1];
  const float* Wk = (const float*)d_in[2];
  const float* Wv = (const float*)d_in[3];
  const float* Wo = (const float*)d_in[4];

  // Workspace layout (bf16, byte offsets), ~72 MB total.
  char* ws = (char*)d_ws;
  const size_t MB = 1024 * 1024;
  bf16* Xb  = (bf16*)(ws);              // 16 MB  (reused as context after QKV)
  bf16* Wqb = (bf16*)(ws + 16 * MB);    //  2 MB
  bf16* Wkb = (bf16*)(ws + 18 * MB);
  bf16* Wvb = (bf16*)(ws + 20 * MB);
  bf16* Wob = (bf16*)(ws + 22 * MB);
  bf16* Qb  = (bf16*)(ws + 24 * MB);    // 16 MB  (B,H,S,Hd)
  bf16* Kb  = (bf16*)(ws + 40 * MB);    // 16 MB  (B,H,S,Hd)
  bf16* Vtb = (bf16*)(ws + 56 * MB);    // 16 MB  (B,H,Hd,S)
  bf16* Ctx = Xb;                       // X is dead after the QKV projections

  const int nX4 = (kM * kD) / 4;        // 2,097,152
  const int nW4 = (kD * kD) / 4;        //   262,144
  cvt_f32_bf16<<<(nX4 + 255) / 256, 256, 0, stream>>>(X,  Xb,  nX4);
  cvt_f32_bf16<<<(nW4 + 255) / 256, 256, 0, stream>>>(Wq, Wqb, nW4);
  cvt_f32_bf16<<<(nW4 + 255) / 256, 256, 0, stream>>>(Wk, Wkb, nW4);
  cvt_f32_bf16<<<(nW4 + 255) / 256, 256, 0, stream>>>(Wv, Wvb, nW4);
  cvt_f32_bf16<<<(nW4 + 255) / 256, 256, 0, stream>>>(Wo, Wob, nW4);

  dim3 ggrid(kM / 128, kD / 128);       // 64 x 8 blocks
  gemm_xwt<1><<<ggrid, 256, 0, stream>>>(Xb, Wqb, (void*)Qb);
  gemm_xwt<1><<<ggrid, 256, 0, stream>>>(Xb, Wkb, (void*)Kb);
  gemm_xwt<2><<<ggrid, 256, 0, stream>>>(Xb, Wvb, (void*)Vtb);

  dim3 fgrid(kS / 128, kB * kH);        // 16 x 64 blocks
  flash_attn<<<fgrid, 256, 0, stream>>>(Qb, Kb, Vtb, Ctx);

  gemm_xwt<3><<<ggrid, 256, 0, stream>>>(Ctx, Wob, d_out);
}
